// GAT_74174085202357
// MI455X (gfx1250) — matrix operand order repaired and checked by reading, not compiled, not run on hardware
//
#include <hip/hip_runtime.h>
#include <cstdint>
#include <cstddef>

typedef __attribute__((ext_vector_type(16))) __bf16 v16bf;
typedef __attribute__((ext_vector_type(8)))  float  v8f;

#define NFEAT   128
#define NHID    16
#define HEADS1  8
#define NCLASS  40
#define HID     128   // NHID*HEADS1
#define L2PAD   96    // cols 0..39 = W2, 48..87 = res2_W, rest zero
#define NEG_SLOPE 0.2f

// ---------- helpers ----------
__device__ __forceinline__ unsigned short f2bf(float f) {
  unsigned int u = __float_as_uint(f);
  u += 0x7FFFu + ((u >> 16) & 1u);          // round-to-nearest-even
  return (unsigned short)(u >> 16);
}
__device__ __forceinline__ unsigned int pack2bf(float lo, float hi) {
  return (unsigned int)f2bf(lo) | ((unsigned int)f2bf(hi) << 16);
}
// monotone uint key for float atomic max
__device__ __forceinline__ unsigned int fkey(float f) {
  unsigned int b = __float_as_uint(f);
  return (b & 0x80000000u) ? ~b : (b | 0x80000000u);
}
__device__ __forceinline__ float funkey(unsigned int k) {
  unsigned int b = (k & 0x80000000u) ? (k & 0x7FFFFFFFu) : ~k;
  return __uint_as_float(b);
}
__device__ __forceinline__ float lrelu(float v) { return v > 0.f ? v : NEG_SLOPE * v; }

union BFrag { v16bf v; unsigned int u[8]; };

__device__ __forceinline__ v16bf load_frag16(const unsigned short* p, int kk, int secondOfs) {
  BFrag F;
  uint4 q0 = *(const uint4*)(p + kk);
  uint4 q1 = *(const uint4*)(p + kk + secondOfs);
  F.u[0] = q0.x; F.u[1] = q0.y; F.u[2] = q0.z; F.u[3] = q0.w;
  F.u[4] = q1.x; F.u[5] = q1.y; F.u[6] = q1.z; F.u[7] = q1.w;
  return F.v;
}

// ---------- utility kernels ----------
__global__ void gat_zero(unsigned int* __restrict__ p, long long n) {
  long long i = (long long)blockIdx.x * blockDim.x + threadIdx.x;
  if (i < n) p[i] = 0u;
}

// f32 -> bf16 vectorized (4 elements / thread)
__global__ void gat_cvt_bf16(const float* __restrict__ in, unsigned int* __restrict__ out,
                             long long n4) {
  long long i = (long long)blockIdx.x * blockDim.x + threadIdx.x;
  if (i >= n4) return;
  float4 f = ((const float4*)in)[i];
  uint2 r;
  r.x = pack2bf(f.x, f.y);
  r.y = pack2bf(f.z, f.w);
  ((uint2*)out)[i] = r;
}

// Build transposed bf16 weight panels:
//  wT1 [256][128]: rows 0..127 = W1^T, rows 128..255 = res1_W^T
//  w2T [ 96][128]: rows 0..39 = W2^T, rows 48..87 = res2_W^T, rest 0
__global__ void gat_prep_w(const float* __restrict__ W1, const float* __restrict__ r1W,
                           const float* __restrict__ W2, const float* __restrict__ r2W,
                           unsigned short* __restrict__ wT1, unsigned short* __restrict__ w2T) {
  int idx = blockIdx.x * blockDim.x + threadIdx.x;
  const int n1 = 256 * 128;
  if (idx < n1) {
    int j = idx >> 7, k = idx & 127;
    float v = (j < 128) ? W1[k * 128 + j] : r1W[k * 128 + (j - 128)];
    wT1[idx] = f2bf(v);
  } else if (idx < n1 + 96 * 128) {
    int t = idx - n1;
    int j = t >> 7, k = t & 127;
    float v = 0.f;
    if (j < NCLASS)              v = W2[k * NCLASS + j];
    else if (j >= 48 && j < 88)  v = r2W[k * NCLASS + (j - 48)];
    w2T[t] = f2bf(v);
  }
}

// ---------- WMMA GEMM 1: [h1 | res1] = xb @ [W1 | res1_W]  (16x64 stripe / wave) ----------
__global__ void gat_gemm1_wmma(const unsigned short* __restrict__ xb,
                               const unsigned short* __restrict__ wT1,
                               float* __restrict__ h1, float* __restrict__ res1, int N) {
  const int lane = threadIdx.x & 31;
  const int half = lane >> 4, lrow = lane & 15;
  const int m0 = blockIdx.x * 16;
  const int nbase = blockIdx.y * 64;     // 4 column tiles per wave
  int ar = m0 + lrow; if (ar > N - 1) ar = N - 1;
  // A fragment rows: lane<16 -> K {kk..kk+15}; lane>=16 -> K {kk+16..kk+31}
  const unsigned short* arow = xb + (size_t)ar * NFEAT + half * 16;
  // B fragment rows: lane<16 -> K {kk..kk+15}; lane>=16 -> K {kk+16..kk+31}
  const unsigned short* b0 = wT1 + (size_t)(nbase +  0 + lrow) * NFEAT + half * 16;
  const unsigned short* b1 = wT1 + (size_t)(nbase + 16 + lrow) * NFEAT + half * 16;
  const unsigned short* b2 = wT1 + (size_t)(nbase + 32 + lrow) * NFEAT + half * 16;
  const unsigned short* b3 = wT1 + (size_t)(nbase + 48 + lrow) * NFEAT + half * 16;
  v8f c0 = {}, c1 = {}, c2 = {}, c3 = {};
#pragma unroll
  for (int kk = 0; kk < NFEAT; kk += 32) {
    v16bf A = load_frag16(arow, kk, 8);    // two halves 8 apart
    c0 = __builtin_amdgcn_wmma_f32_16x16x32_bf16(false, A, false, load_frag16(b0, kk, 8), (short)0, c0, false, false);
    c1 = __builtin_amdgcn_wmma_f32_16x16x32_bf16(false, A, false, load_frag16(b1, kk, 8), (short)0, c1, false, false);
    c2 = __builtin_amdgcn_wmma_f32_16x16x32_bf16(false, A, false, load_frag16(b2, kk, 8), (short)0, c2, false, false);
    c3 = __builtin_amdgcn_wmma_f32_16x16x32_bf16(false, A, false, load_frag16(b3, kk, 8), (short)0, c3, false, false);
  }
  const int row0 = m0 + half * 8;
  v8f cs[4] = {c0, c1, c2, c3};
#pragma unroll
  for (int t = 0; t < 4; ++t) {
    int n0 = nbase + t * 16;
    float* out; int col;
    if (n0 < HID) { out = h1;   col = n0 + lrow; }
    else          { out = res1; col = n0 - HID + lrow; }
#pragma unroll
    for (int r = 0; r < 8; ++r) {
      int row = row0 + r;
      if (row < N) out[(size_t)row * HID + col] = cs[t][r];
    }
  }
}

// ---------- WMMA GEMM 2: h2[N][96] = hb @ [W2 | pad | res2_W | pad]  (16x48 / wave) ----------
__global__ void gat_gemm2_wmma(const unsigned short* __restrict__ hb,
                               const unsigned short* __restrict__ w2T,
                               float* __restrict__ h2, int N) {
  const int lane = threadIdx.x & 31;
  const int half = lane >> 4, lrow = lane & 15;
  const int m0 = blockIdx.x * 16;
  const int nbase = blockIdx.y * 48;     // 3 column tiles per wave
  int ar = m0 + lrow; if (ar > N - 1) ar = N - 1;
  const unsigned short* arow = hb + (size_t)ar * HID + half * 16;
  const unsigned short* b0 = w2T + (size_t)(nbase +  0 + lrow) * HID + half * 16;
  const unsigned short* b1 = w2T + (size_t)(nbase + 16 + lrow) * HID + half * 16;
  const unsigned short* b2 = w2T + (size_t)(nbase + 32 + lrow) * HID + half * 16;
  v8f c0 = {}, c1 = {}, c2 = {};
#pragma unroll
  for (int kk = 0; kk < HID; kk += 32) {
    v16bf A = load_frag16(arow, kk, 8);
    c0 = __builtin_amdgcn_wmma_f32_16x16x32_bf16(false, A, false, load_frag16(b0, kk, 8), (short)0, c0, false, false);
    c1 = __builtin_amdgcn_wmma_f32_16x16x32_bf16(false, A, false, load_frag16(b1, kk, 8), (short)0, c1, false, false);
    c2 = __builtin_amdgcn_wmma_f32_16x16x32_bf16(false, A, false, load_frag16(b2, kk, 8), (short)0, c2, false, false);
  }
  const int row0 = m0 + half * 8;
  v8f cs[3] = {c0, c1, c2};
#pragma unroll
  for (int t = 0; t < 3; ++t) {
    int col = nbase + t * 16 + lrow;
#pragma unroll
    for (int r = 0; r < 8; ++r) {
      int row = row0 + r;
      if (row < N) h2[(size_t)row * L2PAD + col] = cs[t][r];
    }
  }
}

// ---------- attention coefficients ----------
__global__ void gat_attn1(const float* __restrict__ h1,
                          const float* __restrict__ att_src, const float* __restrict__ att_dst,
                          float* __restrict__ a_src, float* __restrict__ a_dst, int N) {
  int i = blockIdx.x * blockDim.x + threadIdx.x;      // over N*8
  if (i >= N * HEADS1) return;
  int n = i >> 3, h = i & 7;
  const float* row = h1 + (size_t)n * HID + h * NHID;
  const float* as = att_src + h * NHID;
  const float* ad = att_dst + h * NHID;
  float s = 0.f, d = 0.f;
#pragma unroll
  for (int c = 0; c < NHID; ++c) { s += row[c] * as[c]; d += row[c] * ad[c]; }
  a_src[i] = s; a_dst[i] = d;
}

__global__ void gat_attn2(const float* __restrict__ h2,
                          const float* __restrict__ att_src, const float* __restrict__ att_dst,
                          float* __restrict__ a_src, float* __restrict__ a_dst, int N) {
  int n = blockIdx.x * blockDim.x + threadIdx.x;
  if (n >= N) return;
  const float* row = h2 + (size_t)n * L2PAD;
  float s = 0.f, d = 0.f;
#pragma unroll
  for (int c = 0; c < NCLASS; ++c) { s += row[c] * att_src[c]; d += row[c] * att_dst[c]; }
  a_src[n] = s; a_dst[n] = d;
}

// ---------- layer 1 edge passes (8 heads) ----------
__device__ __forceinline__ void edge_sd(const int* ei, int E, int e, int& s, int& d) {
  if (e < E) { s = ei[e]; d = ei[E + e]; } else { s = e - E; d = e - E; }
}

__global__ void gat_e1_max(const int* __restrict__ ei, int E, int N,
                           const float* __restrict__ as, const float* __restrict__ ad,
                           unsigned int* __restrict__ emax) {
  int e = blockIdx.x * blockDim.x + threadIdx.x;
  if (e >= E + N) return;
  int s, d; edge_sd(ei, E, e, s, d);
  const float* ps = as + (size_t)s * HEADS1;
  const float* pd = ad + (size_t)d * HEADS1;
#pragma unroll
  for (int h = 0; h < HEADS1; ++h)
    atomicMax(&emax[(size_t)d * HEADS1 + h], fkey(lrelu(ps[h] + pd[h])));
}

__global__ void gat_e1_sum(const int* __restrict__ ei, int E, int N,
                           const float* __restrict__ as, const float* __restrict__ ad,
                           const unsigned int* __restrict__ emax, float* __restrict__ denom) {
  int e = blockIdx.x * blockDim.x + threadIdx.x;
  if (e >= E + N) return;
  int s, d; edge_sd(ei, E, e, s, d);
  const float* ps = as + (size_t)s * HEADS1;
  const float* pd = ad + (size_t)d * HEADS1;
#pragma unroll
  for (int h = 0; h < HEADS1; ++h) {
    float v = lrelu(ps[h] + pd[h]);
    atomicAdd(&denom[(size_t)d * HEADS1 + h], __expf(v - funkey(emax[(size_t)d * HEADS1 + h])));
  }
}

__global__ void gat_e1_scatter(const int* __restrict__ ei, int E, int N,
                               const float* __restrict__ as, const float* __restrict__ ad,
                               const unsigned int* __restrict__ emax, const float* __restrict__ denom,
                               const float* __restrict__ h1, float* __restrict__ acc1) {
  int idx = blockIdx.x * blockDim.x + threadIdx.x;   // over (E+N)*8
  if (idx >= (E + N) * HEADS1) return;
  int e = idx >> 3, h = idx & 7;
  int s, d; edge_sd(ei, E, e, s, d);
  const float* srcrow = h1 + (size_t)s * HID + h * NHID;
  __builtin_prefetch(srcrow, 0, 0);
  float v = lrelu(as[(size_t)s * HEADS1 + h] + ad[(size_t)d * HEADS1 + h]);
  size_t di = (size_t)d * HEADS1 + h;
  float alpha = __expf(v - funkey(emax[di])) / denom[di];
  float* dstp = acc1 + (size_t)d * HID + h * NHID;
  const float4* s4 = (const float4*)srcrow;
#pragma unroll
  for (int j = 0; j < 4; ++j) {
    float4 q = s4[j];
    atomicAdd(dstp + 4 * j + 0, q.x * alpha);
    atomicAdd(dstp + 4 * j + 1, q.y * alpha);
    atomicAdd(dstp + 4 * j + 2, q.z * alpha);
    atomicAdd(dstp + 4 * j + 3, q.w * alpha);
  }
}

// h = elu(acc1 + bias1) + res1 + res1_b ; store bf16 for layer-2 WMMA
__global__ void gat_node1(const float* __restrict__ acc1, const float* __restrict__ res1,
                          const float* __restrict__ bias1, const float* __restrict__ res1_b,
                          unsigned short* __restrict__ hb, int N) {
  int i = blockIdx.x * blockDim.x + threadIdx.x;    // over N*128
  if (i >= N * HID) return;
  int c = i & 127;
  float v = acc1[i] + bias1[c];
  v = v > 0.f ? v : (__expf(v) - 1.0f);
  v += res1[i] + res1_b[c];
  hb[i] = f2bf(v);
}

// ---------- layer 2 edge passes (1 head, 40 ch) ----------
__global__ void gat_e2_max(const int* __restrict__ ei, int E, int N,
                           const float* __restrict__ as, const float* __restrict__ ad,
                           unsigned int* __restrict__ emax) {
  int e = blockIdx.x * blockDim.x + threadIdx.x;
  if (e >= E + N) return;
  int s, d; edge_sd(ei, E, e, s, d);
  atomicMax(&emax[d], fkey(lrelu(as[s] + ad[d])));
}

__global__ void gat_e2_sum(const int* __restrict__ ei, int E, int N,
                           const float* __restrict__ as, const float* __restrict__ ad,
                           const unsigned int* __restrict__ emax, float* __restrict__ denom) {
  int e = blockIdx.x * blockDim.x + threadIdx.x;
  if (e >= E + N) return;
  int s, d; edge_sd(ei, E, e, s, d);
  float v = lrelu(as[s] + ad[d]);
  atomicAdd(&denom[d], __expf(v - funkey(emax[d])));
}

__global__ void gat_e2_scatter(const int* __restrict__ ei, int E, int N,
                               const float* __restrict__ as, const float* __restrict__ ad,
                               const unsigned int* __restrict__ emax, const float* __restrict__ denom,
                               const float* __restrict__ h2, float* __restrict__ acc2) {
  int idx = blockIdx.x * blockDim.x + threadIdx.x;   // over (E+N)*5 groups of 8 cols
  if (idx >= (E + N) * 5) return;
  int e = idx / 5, g = idx - 5 * e;
  int s, d; edge_sd(ei, E, e, s, d);
  const float* srcrow = h2 + (size_t)s * L2PAD + g * 8;
  __builtin_prefetch(srcrow, 0, 0);
  float v = lrelu(as[s] + ad[d]);
  float alpha = __expf(v - funkey(emax[d])) / denom[d];
  float* dstp = acc2 + (size_t)d * NCLASS + g * 8;
#pragma unroll
  for (int j = 0; j < 8; ++j) atomicAdd(dstp + j, srcrow[j] * alpha);
}

// out = log_softmax(acc2 + bias2 + h2[:,48:88] + res2_b)
__global__ void gat_final(const float* __restrict__ acc2, const float* __restrict__ h2,
                          const float* __restrict__ bias2, const float* __restrict__ res2_b,
                          float* __restrict__ out, int N) {
  int n = blockIdx.x * blockDim.x + threadIdx.x;
  if (n >= N) return;
  float v[NCLASS];
  float mx = -3.4e38f;
#pragma unroll
  for (int c = 0; c < NCLASS; ++c) {
    float t = acc2[(size_t)n * NCLASS + c] + bias2[c] + h2[(size_t)n * L2PAD + 48 + c] + res2_b[c];
    v[c] = t;
    mx = fmaxf(mx, t);
  }
  float sum = 0.f;
#pragma unroll
  for (int c = 0; c < NCLASS; ++c) sum += __expf(v[c] - mx);
  float lse = mx + __logf(sum);
#pragma unroll
  for (int c = 0; c < NCLASS; ++c) out[(size_t)n * NCLASS + c] = v[c] - lse;
}

// ---------- launcher ----------
extern "C" void kernel_launch(void* const* d_in, const int* in_sizes, int n_in,
                              void* d_out, int out_size, void* d_ws, size_t ws_size,
                              hipStream_t stream) {
  (void)n_in; (void)out_size; (void)ws_size;
  const float* x        = (const float*)d_in[0];
  const int*   ei       = (const int*)d_in[1];
  const float* W1       = (const float*)d_in[2];
  const float* att_src1 = (const float*)d_in[3];
  const float* att_dst1 = (const float*)d_in[4];
  const float* bias1    = (const float*)d_in[5];
  const float* W2       = (const float*)d_in[6];
  const float* att_src2 = (const float*)d_in[7];
  const float* att_dst2 = (const float*)d_in[8];
  const float* bias2    = (const float*)d_in[9];
  const float* res1_W   = (const float*)d_in[10];
  const float* res1_b   = (const float*)d_in[11];
  const float* res2_W   = (const float*)d_in[12];
  const float* res2_b   = (const float*)d_in[13];
  float* out = (float*)d_out;

  const int N  = in_sizes[0] / NFEAT;
  const int E  = in_sizes[1] / 2;
  const int EE = E + N;

  char* w = (char*)d_ws;
  size_t off = 0;
  auto alloc = [&](size_t bytes) -> void* {
    void* p = w + off;
    off = (off + bytes + 255) & ~(size_t)255;
    return p;
  };
  float*          h1     = (float*)alloc((size_t)N * HID * 4);
  float*          res1   = (float*)alloc((size_t)N * HID * 4);
  float*          acc1   = (float*)alloc((size_t)N * HID * 4);
  float*          asrc1  = (float*)alloc((size_t)N * HEADS1 * 4);
  float*          adst1  = (float*)alloc((size_t)N * HEADS1 * 4);
  unsigned int*   emax1  = (unsigned int*)alloc((size_t)N * HEADS1 * 4);
  float*          denom1 = (float*)alloc((size_t)N * HEADS1 * 4);
  unsigned short* xb     = (unsigned short*)alloc((size_t)N * NFEAT * 2);
  unsigned short* hb     = (unsigned short*)alloc((size_t)N * HID * 2);
  float*          h2     = (float*)alloc((size_t)N * L2PAD * 4);
  float*          acc2   = (float*)alloc((size_t)N * NCLASS * 4);
  float*          asrc2  = (float*)alloc((size_t)N * 4);
  float*          adst2  = (float*)alloc((size_t)N * 4);
  unsigned int*   emax2  = (unsigned int*)alloc((size_t)N * 4);
  float*          denom2 = (float*)alloc((size_t)N * 4);
  unsigned short* wT1    = (unsigned short*)alloc((size_t)256 * 128 * 2);
  unsigned short* w2T    = (unsigned short*)alloc((size_t)96 * 128 * 2);

  const int TB = 256;
  auto nb = [&](long long n) { return dim3((unsigned)((n + TB - 1) / TB)); };

  gat_prep_w<<<nb(256 * 128 + 96 * 128), TB, 0, stream>>>(W1, res1_W, W2, res2_W, wT1, w2T);
  gat_cvt_bf16<<<nb((long long)N * NFEAT / 4), TB, 0, stream>>>(x, (unsigned int*)xb,
                                                                (long long)N * NFEAT / 4);

  gat_zero<<<nb((long long)N * HEADS1), TB, 0, stream>>>(emax1, (long long)N * HEADS1);
  gat_zero<<<nb((long long)N * HEADS1), TB, 0, stream>>>((unsigned int*)denom1, (long long)N * HEADS1);
  gat_zero<<<nb((long long)N * HID),    TB, 0, stream>>>((unsigned int*)acc1, (long long)N * HID);
  gat_zero<<<nb(N),                     TB, 0, stream>>>(emax2, N);
  gat_zero<<<nb(N),                     TB, 0, stream>>>((unsigned int*)denom2, N);
  gat_zero<<<nb((long long)N * NCLASS), TB, 0, stream>>>((unsigned int*)acc2, (long long)N * NCLASS);

  dim3 g1((N + 15) / 16, 4);          // 4 waves-wide column stripes of 64: [W1 | res1_W]
  gat_gemm1_wmma<<<g1, 32, 0, stream>>>(xb, wT1, h1, res1, N);

  gat_attn1<<<nb((long long)N * HEADS1), TB, 0, stream>>>(h1, att_src1, att_dst1, asrc1, adst1, N);

  gat_e1_max    <<<nb(EE), TB, 0, stream>>>(ei, E, N, asrc1, adst1, emax1);
  gat_e1_sum    <<<nb(EE), TB, 0, stream>>>(ei, E, N, asrc1, adst1, emax1, denom1);
  gat_e1_scatter<<<nb((long long)EE * HEADS1), TB, 0, stream>>>(ei, E, N, asrc1, adst1, emax1, denom1, h1, acc1);

  gat_node1<<<nb((long long)N * HID), TB, 0, stream>>>(acc1, res1, bias1, res1_b, hb, N);

  dim3 g2((N + 15) / 16, 2);          // 2 stripes of 48: [W2 | pad | res2_W | pad]
  gat_gemm2_wmma<<<g2, 32, 0, stream>>>(hb, w2T, h2, N);

  gat_attn2<<<nb(N), TB, 0, stream>>>(h2, att_src2, att_dst2, asrc2, adst2, N);

  gat_e2_max    <<<nb(EE), TB, 0, stream>>>(ei, E, N, asrc2, adst2, emax2);
  gat_e2_sum    <<<nb(EE), TB, 0, stream>>>(ei, E, N, asrc2, adst2, emax2, denom2);
  gat_e2_scatter<<<nb((long long)EE * 5), TB, 0, stream>>>(ei, E, N, asrc2, adst2, emax2, denom2, h2, acc2);

  gat_final<<<nb(N), TB, 0, stream>>>(acc2, h2, bias2, res2_b, out, N);
}
